// NCP_LNN_4604204941596
// MI455X (gfx1250) — matrix operand order repaired and checked by reading, not compiled, not run on hardware
//
#include <hip/hip_runtime.h>
#include <hip/hip_bf16.h>
#include <math.h>

#define HID   2048
#define NB    16
#define SEQ   512
#define MROWS (NB*SEQ)   // 8192

#define KITERS        (HID / 32)    // 64 WMMA K-steps
#define CHUNK_COLS    512           // K columns staged per LDS chunk
#define KITER_PER_CH  (CHUNK_COLS / 32)  // 16
#define CHUNKS        (HID / CHUNK_COLS) // 4
#define ROWB          (CHUNK_COLS * 2 + 16) // padded LDS row stride (bank-conflict-free)
#define BUFB          (16 * ROWB)   // bytes per A buffer (16 rows)

typedef __attribute__((ext_vector_type(16))) __bf16 v16bf;
typedef __attribute__((ext_vector_type(8)))  float  v8f;
typedef __attribute__((ext_vector_type(4)))  int    v4i_t;

union Frag16 { v16bf v; uint4 q[2]; };

// ---------------- async global->LDS staging (CDNA5 path) ----------------
#if __has_builtin(__builtin_amdgcn_global_load_async_to_lds_b128) && \
    __has_builtin(__builtin_amdgcn_s_wait_asynccnt)
#define HAVE_ASYNC_LDS 1
#endif

// Stage one 16-row x 512-col bf16 A chunk into LDS (block-cooperative, 128 thr).
// Thread i copies 128B: row = i/8, columns (i%8)*64 .. +63.
__device__ __forceinline__ void stage_A_chunk(const __bf16* __restrict__ A,
                                              char* smem, int buf, int k0) {
  const int tid = threadIdx.x & 127;
  const int r   = tid >> 3;
  const int seg = (tid & 7) * 64;
  const __bf16* g = A + (size_t)r * HID + k0 + seg;
  char* l = smem + buf * BUFB + r * ROWB + seg * 2;
#if HAVE_ASYNC_LDS
#pragma unroll
  for (int i = 0; i < 8; ++i)
    __builtin_amdgcn_global_load_async_to_lds_b128(
        (v4i_t*)(g + i * 8), (v4i_t*)(l + i * 16), 0, 0);
#else
#pragma unroll
  for (int i = 0; i < 8; ++i)
    *(uint4*)(l + i * 16) = *(const uint4*)(g + i * 8);
#endif
}

__device__ __forceinline__ void wait_stage() {
#if HAVE_ASYNC_LDS
  __builtin_amdgcn_s_wait_asynccnt(0);
#endif
  __syncthreads();
}

// ---------------- pipelined 16-row GEMM core ----------------
// acc[NT] += A(16 x HID, row-major) * B-tiles nt0..nt0+NT-1 from packed BP.
// A is staged via async double-buffered LDS chunks; B is register double-buffered
// so WMMAs overlap with the next tile's global loads.
template <int NT>
__device__ __forceinline__ void gemm_rows16(const __bf16* __restrict__ A,
                                            const __bf16* __restrict__ BP,
                                            int nt0, char* smem, v8f acc[NT]) {
  const int lane = threadIdx.x & 31;
  const int m    = lane & 15;
  const int kgrp = lane >> 4;
  const __bf16* bbase = BP + (size_t)lane * 16;

  stage_A_chunk(A, smem, 0, 0);
  wait_stage();
  stage_A_chunk(A, smem, 1, CHUNK_COLS);   // prefetch chunk 1 while computing chunk 0

  Frag16 bc[NT], bn[NT];
#pragma unroll
  for (int t = 0; t < NT; ++t) {
    const __bf16* bp = bbase + ((size_t)nt0 + t) * 512;
    bc[t].q[0] = *(const uint4*)bp;
    bc[t].q[1] = *(const uint4*)(bp + 8);
  }

  for (int cc = 0; cc < CHUNKS; ++cc) {
    const int buf = cc & 1;
#pragma unroll
    for (int kkl = 0; kkl < KITER_PER_CH; ++kkl) {
      const int kk = cc * KITER_PER_CH + kkl;

      // A fragment from LDS (padded rows -> conflict-free ds_load_b128)
      Frag16 a;
      const char* ab = smem + buf * BUFB + m * ROWB + (size_t)(kkl * 32 + kgrp * 16) * 2;
      a.q[0] = *(const uint4*)ab;
      a.q[1] = *(const uint4*)(ab + 16);

      // preload next K-step's B fragments (keeps loads in flight across the WMMAs)
      if (kk + 1 < KITERS) {
        const __bf16* bpn = bbase + ((size_t)(kk + 1) * 128 + nt0) * 512;
#pragma unroll
        for (int t = 0; t < NT; ++t) {
          bn[t].q[0] = *(const uint4*)(bpn + (size_t)t * 512);
          bn[t].q[1] = *(const uint4*)(bpn + (size_t)t * 512 + 8);
        }
        __builtin_prefetch((const void*)(bpn + (size_t)128 * 512), 0, 3);
      }

#pragma unroll
      for (int t = 0; t < NT; ++t)
        acc[t] = __builtin_amdgcn_wmma_f32_16x16x32_bf16(
            false, a.v, false, bc[t].v, (short)0, acc[t], false, false);

      if (kk + 1 < KITERS) {
#pragma unroll
        for (int t = 0; t < NT; ++t) bc[t] = bn[t];
      }
    }
    if (cc + 1 < CHUNKS) {
      wait_stage();                                   // chunk cc+1 ready, buf cc free
      if (cc + 2 < CHUNKS) stage_A_chunk(A, smem, cc & 1, (cc + 2) * CHUNK_COLS);
    }
  }
}

// ---------------- pack / prep kernels ----------------

__global__ void k_pack_x(const float* __restrict__ x, __bf16* __restrict__ xh, int n) {
  int i = blockIdx.x * blockDim.x + threadIdx.x;
  if (i < n) xh[i] = (__bf16)x[i];
}

// Pack W[n][k] into fragment-ready layout for V_WMMA B-operands (32x16 KxN tiles):
//   WP[(ktile*128 + ntile)*512 + lane*16 + j] = W[ntile*16+(lane&15)][ktile*32+(lane>>4)*16+j]
__global__ void k_pack_w(const float* __restrict__ W, const float* __restrict__ mask,
                         __bf16* __restrict__ WP) {
  int i = blockIdx.x * blockDim.x + threadIdx.x;
  if (i >= HID * HID) return;
  int j     = i & 15;
  int lane  = (i >> 4) & 31;
  int tile  = i >> 9;
  int ntile = tile & 127;   // HID/16
  int ktile = tile >> 7;    // HID/32
  int kgrp  = lane >> 4;
  int n = ntile * 16 + (lane & 15);
  int k = ktile * 32 + kgrp * 16 + j;
  float v = W[(size_t)n * HID + k];
  if (mask) v *= mask[(size_t)n * HID + k];
  WP[i] = (__bf16)v;
}

__global__ void k_alpha(const float* __restrict__ tau, float* __restrict__ alpha) {
  int i = blockIdx.x * blockDim.x + threadIdx.x;
  if (i < HID) {
    float tc = fminf(20.0f, fmaxf(1.0f, tau[i]));
    float a  = 0.5f / tc;
    alpha[i] = fminf(1.0f, fmaxf(0.0f, a));
  }
}

__global__ void k_zero_state(float* __restrict__ sf, __bf16* __restrict__ sb, int n) {
  int i = blockIdx.x * blockDim.x + threadIdx.x;
  if (i < n) { sf[i] = 0.0f; sb[i] = (__bf16)0.0f; }
}

// ---------------- input projection GEMM ----------------
// inp[8192 x 2048] = xh * Wi^T + b. block = 128 (4 waves), wave = 16x64.
// grid = (HID/256, MROWS/16) = (8, 512)
__global__ void __launch_bounds__(128)
k_gemm_input(const __bf16* __restrict__ xh, const __bf16* __restrict__ WiP,
             const float* __restrict__ bias, float* __restrict__ inp) {
  __shared__ char smem[2 * BUFB];
  const int lane = threadIdx.x & 31;
  const int wave = threadIdx.x >> 5;
  const int mt   = blockIdx.y;
  const int nt0  = blockIdx.x * 16 + wave * 4;
  const int kgrp = lane >> 4;

  v8f acc[4] = {};
  gemm_rows16<4>(xh + (size_t)(mt * 16) * HID, WiP, nt0, smem, acc);

  const int n_in = lane & 15;
  float* orow = inp + (size_t)(mt * 16) * HID;
#pragma unroll
  for (int r = 0; r < 8; ++r) {
    int mm = kgrp * 8 + r;
    size_t ro = (size_t)mm * HID;
#pragma unroll
    for (int t = 0; t < 4; ++t) {
      int n = (nt0 + t) * 16 + n_in;
      orow[ro + n] = acc[t][r] + bias[n];
    }
  }
}

// ---------------- recurrent step ----------------
// target = tanh(inp_t + state @ Wm^T); state = clip(state + alpha*(target-state), -1, 1)
// block = 128 (4 waves), wave = 16x32 (2 N-tiles). grid = HID/128 = 16.
__global__ void __launch_bounds__(128)
k_step(const float* __restrict__ inp, const __bf16* __restrict__ WmP,
       const float* __restrict__ alpha,
       const float* __restrict__ sfin, const __bf16* __restrict__ sbin,
       float* __restrict__ sfout, __bf16* __restrict__ sbout,
       float* __restrict__ outs, int t) {
  __shared__ char smem[2 * BUFB];
  const int lane = threadIdx.x & 31;
  const int wave = threadIdx.x >> 5;
  const int nt0  = blockIdx.x * 8 + wave * 2;
  const int kgrp = lane >> 4;

  v8f acc[2] = {};
  gemm_rows16<2>(sbin, WmP, nt0, smem, acc);

  const int n_in = lane & 15;
#pragma unroll
  for (int r = 0; r < 8; ++r) {
    int b = kgrp * 8 + r;   // batch row
#pragma unroll
    for (int tt = 0; tt < 2; ++tt) {
      int n = (nt0 + tt) * 16 + n_in;
      float pre = acc[tt][r] + inp[((size_t)b * SEQ + t) * HID + n];
      float tg  = tanhf(pre);
      float so  = sfin[(size_t)b * HID + n];
      float sn  = so + alpha[n] * (tg - so);
      sn = fminf(1.0f, fmaxf(-1.0f, sn));
      sfout[(size_t)b * HID + n] = sn;
      sbout[(size_t)b * HID + n] = (__bf16)sn;
      outs[((size_t)b * SEQ + t) * HID + n] = sn;
    }
  }
}

// ---------------- LayerNorm (in-place on d_out) ----------------
__global__ void k_layernorm(float* __restrict__ out, const float* __restrict__ gamma,
                            const float* __restrict__ beta) {
  __shared__ float ssum[256];
  __shared__ float ssq[256];
  const int row = blockIdx.x;
  float* p = out + (size_t)row * HID;
  float v[8];
  float s = 0.0f, q = 0.0f;
#pragma unroll
  for (int i = 0; i < 8; ++i) {
    v[i] = p[threadIdx.x + i * 256];
    s += v[i];
    q += v[i] * v[i];
  }
  ssum[threadIdx.x] = s;
  ssq[threadIdx.x]  = q;
  __syncthreads();
  for (int off = 128; off > 0; off >>= 1) {
    if ((int)threadIdx.x < off) {
      ssum[threadIdx.x] += ssum[threadIdx.x + off];
      ssq[threadIdx.x]  += ssq[threadIdx.x + off];
    }
    __syncthreads();
  }
  float mu  = ssum[0] * (1.0f / HID);
  float var = ssq[0] * (1.0f / HID) - mu * mu;
  float rs  = rsqrtf(var + 1e-5f);
#pragma unroll
  for (int i = 0; i < 8; ++i) {
    int c = threadIdx.x + i * 256;
    p[c] = (v[i] - mu) * rs * gamma[c] + beta[c];
  }
}

// ---------------- host launcher ----------------
extern "C" void kernel_launch(void* const* d_in, const int* in_sizes, int n_in,
                              void* d_out, int out_size, void* d_ws, size_t ws_size,
                              hipStream_t stream) {
  (void)in_sizes; (void)n_in; (void)out_size; (void)ws_size;
  const float* x     = (const float*)d_in[0];
  const float* Wi    = (const float*)d_in[1];
  const float* bias  = (const float*)d_in[2];
  const float* Wr    = (const float*)d_in[3];
  const float* mask  = (const float*)d_in[4];
  const float* tau   = (const float*)d_in[5];
  const float* gamma = (const float*)d_in[6];
  const float* beta  = (const float*)d_in[7];
  float* out = (float*)d_out;

  char* ws = (char*)d_ws;
  size_t off = 0;
  auto take = [&](size_t bytes) {
    char* p = ws + off;
    off = (off + bytes + 255) & ~(size_t)255;
    return p;
  };
  __bf16* xh    = (__bf16*)take((size_t)MROWS * HID * 2);  // 32 MB
  __bf16* WiP   = (__bf16*)take((size_t)HID * HID * 2);    //  8 MB
  __bf16* WmP   = (__bf16*)take((size_t)HID * HID * 2);    //  8 MB
  float*  inp   = (float*) take((size_t)MROWS * HID * 4);  // 64 MB
  float*  alpha = (float*) take((size_t)HID * 4);
  float*  sf    = (float*) take((size_t)2 * NB * HID * 4); // ping-pong f32 state
  __bf16* sb    = (__bf16*)take((size_t)2 * NB * HID * 2); // ping-pong bf16 state

  {
    int n = MROWS * HID;
    k_pack_x<<<dim3((n + 255) / 256), dim3(256), 0, stream>>>(x, xh, n);
  }
  k_pack_w<<<dim3((HID * HID) / 256), dim3(256), 0, stream>>>(Wi, nullptr, WiP);
  k_pack_w<<<dim3((HID * HID) / 256), dim3(256), 0, stream>>>(Wr, mask, WmP);
  k_alpha<<<dim3((HID + 255) / 256), dim3(256), 0, stream>>>(tau, alpha);
  k_zero_state<<<dim3((2 * NB * HID + 255) / 256), dim3(256), 0, stream>>>(sf, sb, 2 * NB * HID);

  // input projection
  k_gemm_input<<<dim3(HID / 256, MROWS / 16), dim3(128), 0, stream>>>(xh, WiP, bias, inp);

  // recurrent scan (512 sequential steps; packed Wm stays L2-resident)
  for (int t = 0; t < SEQ; ++t) {
    int pi = t & 1, po = (t + 1) & 1;
    k_step<<<dim3(HID / 128), dim3(128), 0, stream>>>(
        inp, WmP, alpha,
        sf + (size_t)pi * NB * HID, sb + (size_t)pi * NB * HID,
        sf + (size_t)po * NB * HID, sb + (size_t)po * NB * HID,
        out, t);
  }

  k_layernorm<<<dim3(MROWS), dim3(256), 0, stream>>>(out, gamma, beta);
}
